// PixelSharp_78958678769719
// MI455X (gfx1250) — compile-verified, matrix operand order checked by reading, not run on hardware
//
#include <hip/hip_runtime.h>
#include <cstdint>
#include <cstddef>

// ---------------------------------------------------------------------------
// PixelCNN-style forward for MI455X (gfx1250, wave32, WMMA bf16).
// All channel contractions are v_wmma_f32_16x16x32_bf16 GEMMs over the 32768
// spatial columns. B-operands (activations, with the masked-conv spatial
// shifts folded in) are staged through LDS with double-buffered
// global_load_async_to_lds_b128 (ASYNCcnt), overlapping the next K-step's
// gather with the current WMMAs.
// ---------------------------------------------------------------------------

#define S_TOT 32768   // N*H*W = 32*32*32
#define CS    160     // channel stride (144 padded to multiple of 32)

typedef __attribute__((ext_vector_type(16))) __bf16 v16bf;
typedef __attribute__((ext_vector_type(8)))  float  v8f;

union FragB { uint4 u[2]; v16bf v; };

__device__ __forceinline__ float bf2f(unsigned short u) {
  return __uint_as_float(((unsigned)u) << 16);
}
__device__ __forceinline__ unsigned short f2bf(float f) {
  unsigned u = __float_as_uint(f);
  u += 0x7fffu + ((u >> 16) & 1u);            // round-to-nearest-even
  return (unsigned short)(u >> 16);
}

// One lane copies 64B (one column's 32 bf16 K-values) global -> LDS.
// Per ISA 10.4/15.18: INST_OFFSET is added to BOTH the global and LDS address.
__device__ __forceinline__ void async_col64(const void* g, unsigned ldsoff) {
  asm volatile("global_load_async_to_lds_b128 %0, %1, off"
               :: "v"(ldsoff), "v"(g) : "memory");
  asm volatile("global_load_async_to_lds_b128 %0, %1, off offset:16"
               :: "v"(ldsoff), "v"(g) : "memory");
  asm volatile("global_load_async_to_lds_b128 %0, %1, off offset:32"
               :: "v"(ldsoff), "v"(g) : "memory");
  asm volatile("global_load_async_to_lds_b128 %0, %1, off offset:48"
               :: "v"(ldsoff), "v"(g) : "memory");
}

__device__ __forceinline__ void mask_frag(FragB& f, unsigned m) {
  f.u[0].x &= m; f.u[0].y &= m; f.u[0].z &= m; f.u[0].w &= m;
  f.u[1].x &= m; f.u[1].y &= m; f.u[1].z &= m; f.u[1].w &= m;
}

// ---------------------------------------------------------------------------
// Generic WMMA GEMM:  out[M=48*gridDim.y][S] = W[M][Ktot] * B(panels) + bias
// Each panel supplies kcols channels of a [S][cstride] bf16 activation buffer,
// read at spatial offset (dy,dx) with zero padding outside the 32x32 image.
// Wave tile: 32 columns x 48 rows  -> 6 WMMAs per 32-wide K step.
// Epilogue: optional bf16 store, optional f32 accumulate (+bf16 mirror).
// ---------------------------------------------------------------------------
struct Panel { const unsigned short* act; int dy, dx, kcols, cstride; };
struct GemmCfg {
  const unsigned short* W;      // [Mrows][Ktot] bf16 packed
  const float*          bias;   // [Mrows]
  unsigned short*       outbf;  // [S][ostride] bf16 raw output (or null)
  float*                accf;   // [S][ostride] f32 accumulate in/out (or null)
  unsigned short*       accbf;  // bf16 mirror of accf (valid if accf != null)
  int Ktot, npanels, ostride;
  Panel p[6];
};

__global__ __launch_bounds__(128) void k_gemm(GemmCfg cfg) {
  // 4 waves * (double buffer of 32 columns * 64B) = 16 KB
  __shared__ __align__(16) unsigned char ldsb[4 * 4096];

  const int lane = threadIdx.x & 31;
  const int wave = threadIdx.x >> 5;
  const int ln   = lane & 15;     // B column within tile / A row within tile
  const int lh   = lane >> 4;     // K-chunk selector per documented layout
  const int col0 = blockIdx.x * 128 + wave * 32;   // 32 cols = one image row
  const int h    = (col0 >> 5) & 31;               // uniform over the wave
  const int mbase = blockIdx.y * 48;
  unsigned char* myl = ldsb + wave * 4096;

  v8f acc[2][3];
#pragma unroll
  for (int u = 0; u < 2; ++u)
#pragma unroll
    for (int t = 0; t < 3; ++t)
#pragma unroll
      for (int i = 0; i < 8; ++i) acc[u][t][i] = 0.0f;

  int koff = 0;
  for (int pi = 0; pi < cfg.npanels; ++pi) {
    const Panel pp = cfg.p[pi];
    const int hh = h + pp.dy;
    if ((unsigned)hh >= 32u) { koff += pp.kcols; continue; }  // uniform skip

    // loader: this lane owns image column w = lane of the row; clamp OOB
    const bool okl = (unsigned)(lane + pp.dx) < 32u;
    const int  soff = okl ? (pp.dy * 32 + pp.dx) : 0;
    const unsigned short* gcol =
        pp.act + (size_t)(col0 + lane + soff) * pp.cstride;

    // reader zero-masks per column tile (w of tile u = u*16 + ln)
    const unsigned msk0 = ((unsigned)(ln + pp.dx) < 32u) ? ~0u : 0u;
    const unsigned msk1 = ((unsigned)(16 + ln + pp.dx) < 32u) ? ~0u : 0u;

    const int nsteps = pp.kcols >> 5;
    // prologue: stage K-step 0 into buffer 0
    async_col64(gcol, (unsigned)(uintptr_t)(myl + lane * 64));
    int buf = 0;
    for (int ks = 0; ks < nsteps; ++ks) {
      const int kg = koff + ks * 32 + lh * 8;  // packed-weight column base
      // A fragments (distinct regs per M-tile so loads overlap)
      FragB a0, a1, a2;
      {
        const uint4* q0 = (const uint4*)(cfg.W +
            (size_t)(mbase + 0 * 16 + ln) * cfg.Ktot + kg);
        const uint4* q1 = (const uint4*)(cfg.W +
            (size_t)(mbase + 1 * 16 + ln) * cfg.Ktot + kg);
        const uint4* q2 = (const uint4*)(cfg.W +
            (size_t)(mbase + 2 * 16 + ln) * cfg.Ktot + kg);
        a0.u[0] = q0[0]; a0.u[1] = q0[2];
        a1.u[0] = q1[0]; a1.u[1] = q1[2];
        a2.u[0] = q2[0]; a2.u[1] = q2[2];
      }
      // stage next K-step into the other buffer while we compute
      if (ks + 1 < nsteps) {
        async_col64(gcol + (ks + 1) * 32,
                    (unsigned)(uintptr_t)(myl + (buf ^ 1) * 2048 + lane * 64));
        asm volatile("s_wait_asynccnt 4" ::: "memory");
      } else {
        asm volatile("s_wait_asynccnt 0" ::: "memory");
      }
      // B fragments from LDS (column-major 64B per column)
      FragB b0, b1;
      {
        const uint4* r0 =
            (const uint4*)(myl + buf * 2048 + ln * 64 + lh * 16);
        const uint4* r1 =
            (const uint4*)(myl + buf * 2048 + (16 + ln) * 64 + lh * 16);
        b0.u[0] = r0[0]; b0.u[1] = r0[2];
        b1.u[0] = r1[0]; b1.u[1] = r1[2];
      }
      mask_frag(b0, msk0);
      mask_frag(b1, msk1);
      acc[0][0] = __builtin_amdgcn_wmma_f32_16x16x32_bf16(
          false, a0.v, false, b0.v, (short)0, acc[0][0], false, false);
      acc[0][1] = __builtin_amdgcn_wmma_f32_16x16x32_bf16(
          false, a1.v, false, b0.v, (short)0, acc[0][1], false, false);
      acc[0][2] = __builtin_amdgcn_wmma_f32_16x16x32_bf16(
          false, a2.v, false, b0.v, (short)0, acc[0][2], false, false);
      acc[1][0] = __builtin_amdgcn_wmma_f32_16x16x32_bf16(
          false, a0.v, false, b1.v, (short)0, acc[1][0], false, false);
      acc[1][1] = __builtin_amdgcn_wmma_f32_16x16x32_bf16(
          false, a1.v, false, b1.v, (short)0, acc[1][1], false, false);
      acc[1][2] = __builtin_amdgcn_wmma_f32_16x16x32_bf16(
          false, a2.v, false, b1.v, (short)0, acc[1][2], false, false);
      buf ^= 1;
    }
    koff += pp.kcols;
  }

  // Epilogue. C/D layout: VGPR r, lanes<16 -> M=r, lanes>=16 -> M=8+r.
#pragma unroll
  for (int u = 0; u < 2; ++u) {
    const int s = col0 + u * 16 + ln;
#pragma unroll
    for (int t = 0; t < 3; ++t) {
      const int mo = mbase + t * 16 + lh * 8;   // 8 consecutive channels
      const size_t obase = (size_t)s * cfg.ostride + mo;
      float r[8];
#pragma unroll
      for (int i = 0; i < 8; ++i) r[i] = acc[u][t][i] + cfg.bias[mo + i];
      if (cfg.accf) {
        float* fp = cfg.accf + obase;
#pragma unroll
        for (int i = 0; i < 8; ++i) {
          float nv = fp[i] + r[i];
          fp[i] = nv;
          cfg.accbf[obase + i] = f2bf(nv);
        }
      }
      if (cfg.outbf) {
#pragma unroll
        for (int i = 0; i < 8; ++i) cfg.outbf[obase + i] = f2bf(r[i]);
      }
    }
  }
}

// ---------------------------------------------------------------------------
// Layer-0 prep: build im2col panels for the 3-channel input.
//   Pv0[s][t*3+c] = vx tap (kh,kw) with d=1   (vx = rows shifted down by 1)
//   Ph0[s][c]     = shift_right(hx,1),  Ph0[s][3+c] = hx
// ---------------------------------------------------------------------------
__global__ void k_prep0(const float* __restrict__ x,
                        unsigned short* __restrict__ Pv0,
                        unsigned short* __restrict__ Ph0) {
  int s = blockIdx.x * 256 + threadIdx.x;
  if (s >= S_TOT) return;
  int n = s >> 10, hw = s & 1023, h = hw >> 5, w = hw & 31;
  auto X = [&](int c, int hh, int ww) -> float {
    if ((unsigned)hh >= 32u || (unsigned)ww >= 32u) return 0.0f;
    return 2.0f * x[((n * 3 + c) << 10) + (hh << 5) + ww] - 1.0f;
  };
  for (int kh = 0; kh < 2; ++kh)
    for (int kw = 0; kw < 3; ++kw)
      for (int c = 0; c < 3; ++c)   // vx[h+kh-1] = x[h+kh-2] (row shift)
        Pv0[(size_t)s * 32 + (kh * 3 + kw) * 3 + c] =
            f2bf(X(c, h + kh - 2, w + kw - 1));
  for (int c = 0; c < 3; ++c) {
    Ph0[(size_t)s * 32 + c]     = f2bf(X(c, h, w - 1));
    Ph0[(size_t)s * 32 + 3 + c] = f2bf(X(c, h, w));
  }
}

// Layer-0 residual: VXf = vx2vx(vx) (1x1 conv from 3 channels)
__global__ void k_resvx(const float* __restrict__ x,
                        const float* __restrict__ wv,
                        const float* __restrict__ bv,
                        float* __restrict__ VXf) {
  long long idx = (long long)blockIdx.x * 256 + threadIdx.x;
  if (idx >= (long long)S_TOT * CS) return;
  int m = (int)(idx % CS);
  long long s = idx / CS;
  if (m >= 144) return;
  int n = (int)(s >> 10), hw = (int)(s & 1023), h = hw >> 5, w = hw & 31;
  float acc = bv[m];
  if (h > 0)
    for (int c = 0; c < 3; ++c)
      acc += wv[m * 3 + c] *
             (2.0f * x[((n * 3 + c) << 10) + ((h - 1) << 5) + w] - 1.0f);
  VXf[(size_t)s * CS + m] = acc;
}

// ---------------------------------------------------------------------------
// BatchNorm (training-mode batch stats) over concat_minmax(buf)
// ---------------------------------------------------------------------------
__global__ void k_stats(const unsigned short* __restrict__ buf, int cstride,
                        float* __restrict__ sum, float* __restrict__ sumsq) {
  const int pb = blockIdx.y;                 // channel pair 0..71
  const int s = blockIdx.x * 256 + threadIdx.x;
  float a = bf2f(buf[(size_t)s * cstride + 2 * pb]);
  float b = bf2f(buf[(size_t)s * cstride + 2 * pb + 1]);
  float mx = fmaxf(a, b), mn = fminf(a, b);
  __shared__ float sh0[256], sh1[256], sh2[256], sh3[256];
  int tid = threadIdx.x;
  sh0[tid] = mx; sh1[tid] = mn; sh2[tid] = mx * mx; sh3[tid] = mn * mn;
  __syncthreads();
  for (int st = 128; st > 0; st >>= 1) {
    if (tid < st) {
      sh0[tid] += sh0[tid + st]; sh1[tid] += sh1[tid + st];
      sh2[tid] += sh2[tid + st]; sh3[tid] += sh3[tid + st];
    }
    __syncthreads();
  }
  if (tid == 0) {
    atomicAdd(&sum[2 * pb],     sh0[0]);
    atomicAdd(&sum[2 * pb + 1], sh1[0]);
    atomicAdd(&sumsq[2 * pb],     sh2[0]);
    atomicAdd(&sumsq[2 * pb + 1], sh3[0]);
  }
}

__global__ void k_bnfin(const float* sum, const float* sumsq,
                        const float* g, const float* bb,
                        float* sc, float* bi, float inv_n) {
  int t = threadIdx.x;
  if (t >= 144) return;
  float mean = sum[t] * inv_n;
  float var = sumsq[t] * inv_n - mean * mean;   // biased var
  float s = g[t] * rsqrtf(var + 1e-5f);
  sc[t] = s;
  bi[t] = bb[t] - mean * s;
}

// vx = vx(prev, already in VXf) + BN(minmax(vcx)); refresh bf16 mirror
__global__ void k_vupdate(const unsigned short* __restrict__ vcx,
                          const float* sc, const float* bi,
                          float* __restrict__ VXf,
                          unsigned short* __restrict__ VXbf) {
  long long idx = (long long)blockIdx.x * 256 + threadIdx.x;
  if (idx >= (long long)S_TOT * 72) return;
  int p = (int)(idx % 72);
  long long s = idx / 72;
  float a = bf2f(vcx[s * CS + 2 * p]);
  float b = bf2f(vcx[s * CS + 2 * p + 1]);
  float mx = fmaxf(a, b), mn = fminf(a, b);
  int c0 = 2 * p, c1 = 2 * p + 1;
  float v0 = VXf[s * CS + c0] + sc[c0] * mx + bi[c0];
  float v1 = VXf[s * CS + c1] + sc[c1] * mn + bi[c1];
  VXf[s * CS + c0] = v0;  VXf[s * CS + c1] = v1;
  VXbf[s * CS + c0] = f2bf(v0);  VXbf[s * CS + c1] = f2bf(v1);
}

// h_act = BN(minmax(T)); skip += h_act; HACT = bf16(h_act)
__global__ void k_hupdate(const unsigned short* __restrict__ T,
                          const float* sc, const float* bi,
                          float* __restrict__ SKIPf,
                          unsigned short* __restrict__ HACT) {
  long long idx = (long long)blockIdx.x * 256 + threadIdx.x;
  if (idx >= (long long)S_TOT * 72) return;
  int p = (int)(idx % 72);
  long long s = idx / 72;
  float a = bf2f(T[s * CS + 2 * p]);
  float b = bf2f(T[s * CS + 2 * p + 1]);
  float mx = fmaxf(a, b), mn = fminf(a, b);
  int c0 = 2 * p, c1 = 2 * p + 1;
  float h0 = sc[c0] * mx + bi[c0];
  float h1 = sc[c1] * mn + bi[c1];
  SKIPf[s * CS + c0] += h0;  SKIPf[s * CS + c1] += h1;
  HACT[s * CS + c0] = f2bf(h0);  HACT[s * CS + c1] = f2bf(h1);
}

__global__ void k_skipmm(const float* __restrict__ SKIPf,
                         unsigned short* __restrict__ SKIPbf) {
  long long idx = (long long)blockIdx.x * 256 + threadIdx.x;
  if (idx >= (long long)S_TOT * 72) return;
  int p = (int)(idx % 72);
  long long s = idx / 72;
  float a = SKIPf[s * CS + 2 * p], b = SKIPf[s * CS + 2 * p + 1];
  SKIPbf[s * CS + 2 * p]     = f2bf(fmaxf(a, b));
  SKIPbf[s * CS + 2 * p + 1] = f2bf(fminf(a, b));
}

// final head: minmax(y1) then the 3-row causal 1x1 (dense-zero-padded W2)
__global__ void k_out2(const unsigned short* __restrict__ Y1,
                       const unsigned short* __restrict__ W2,
                       const float* __restrict__ B2,
                       float* __restrict__ out) {
  int s = blockIdx.x * 256 + threadIdx.x;
  if (s >= S_TOT) return;
  float a0 = B2[0], a1 = B2[1], a2 = B2[2];
  for (int j = 0; j < 192; ++j) {
    float a = bf2f(Y1[(size_t)s * 384 + 2 * j]);
    float b = bf2f(Y1[(size_t)s * 384 + 2 * j + 1]);
    float mx = fmaxf(a, b), mn = fminf(a, b);
    a0 += bf2f(W2[2 * j]) * mx       + bf2f(W2[2 * j + 1]) * mn;
    a1 += bf2f(W2[384 + 2 * j]) * mx + bf2f(W2[384 + 2 * j + 1]) * mn;
    a2 += bf2f(W2[768 + 2 * j]) * mx + bf2f(W2[768 + 2 * j + 1]) * mn;
  }
  int n = s >> 10, hw = s & 1023;
  out[n * 3072 + hw]        = a0;
  out[n * 3072 + 1024 + hw] = a1;
  out[n * 3072 + 2048 + hw] = a2;
}

// ---------------------------------------------------------------------------
// Weight packing (f32 params -> packed bf16 GEMM weights; zeros pre-memset)
// ---------------------------------------------------------------------------
__global__ void k_pack_vconv(const float* __restrict__ w,
                             unsigned short* __restrict__ dst,
                             int Cin, int colstride, int Ktot) {
  int idx = blockIdx.x * 256 + threadIdx.x;
  int total = 144 * 6 * Cin;
  if (idx >= total) return;
  int c = idx % Cin;
  int t = (idx / Cin) % 6;
  int m = idx / (Cin * 6);
  int kh = t / 3, kw = t % 3;
  dst[(size_t)m * Ktot + t * colstride + c] =
      f2bf(w[((m * Cin + c) * 2 + kh) * 3 + kw]);
}

__global__ void k_pack_dense(const float* __restrict__ w,
                             unsigned short* __restrict__ dst,
                             int rows, int cols, int Ktot, int koff) {
  int idx = blockIdx.x * 256 + threadIdx.x;
  if (idx >= rows * cols) return;
  int m = idx / cols, c = idx % cols;
  dst[(size_t)m * Ktot + koff + c] = f2bf(w[m * cols + c]);
}

// block-lower-triangular subpixel weights: piece q (rows q*po..) uses first
// in_step*(q+1) input channels
__global__ void k_pack_block(const float* p0, const float* p1, const float* p2,
                             unsigned short* __restrict__ dst,
                             int piece_out, int in_step, int np,
                             int Ktot, int koff) {
  int cw = in_step * np;
  int total = piece_out * np * cw;
  int idx = blockIdx.x * 256 + threadIdx.x;
  if (idx >= total) return;
  int m = idx / cw, c = idx % cw;
  int q = m / piece_out, r = m % piece_out;
  int width = in_step * (q + 1);
  if (c >= width) return;
  const float* p = (q == 0) ? p0 : ((q == 1) ? p1 : p2);
  dst[(size_t)m * Ktot + koff + c] = f2bf(p[r * width + c]);
}

__global__ void k_bias_h(const float* hb, const float* v2hb,
                         const float* p0, const float* p1, const float* p2,
                         float* dst) {
  int m = threadIdx.x;
  if (m >= 144) return;
  int q = m / 48;
  const float* p = (q == 0) ? p0 : ((q == 1) ? p1 : p2);
  dst[m] = hb[m] + v2hb[m] + (p ? p[m % 48] : 0.0f);
}

__global__ void k_bias_block(const float* b0, const float* b1, const float* b2,
                             float* dst, int piece_out, int rows) {
  int m = blockIdx.x * 256 + threadIdx.x;
  if (m >= rows) return;
  int q = m / piece_out;
  const float* p = (q == 0) ? b0 : ((q == 1) ? b1 : b2);
  dst[m] = p[m % piece_out];
}

// ---------------------------------------------------------------------------
extern "C" void kernel_launch(void* const* d_in, const int* in_sizes, int n_in,
                              void* d_out, int out_size, void* d_ws,
                              size_t ws_size, hipStream_t stream) {
  (void)in_sizes; (void)n_in; (void)out_size; (void)ws_size;
  auto F = [&](int i) { return (const float*)d_in[i]; };
  int ip = 0;
  const float* Xin = F(ip++);
  const float* vx2vx_w = F(ip++);
  const float* vx2vx_b = F(ip++);
  struct LP {
    const float *v_w, *v_b, *v_g, *v_bb, *h_w, *h_b;
    const float *hsw[3], *hsb[3];
    const float *h_g, *h_bb, *v2h_w, *v2h_b;
    const float *h2w[3], *h2b[3];
  } L[15];
  for (int i = 0; i < 15; ++i) {
    int ns = (i == 0) ? 2 : 3;
    L[i].v_w = F(ip++); L[i].v_b = F(ip++);
    L[i].v_g = F(ip++); L[i].v_bb = F(ip++);
    L[i].h_w = F(ip++); L[i].h_b = F(ip++);
    for (int j = 0; j < 3; ++j) { L[i].hsw[j] = nullptr; L[i].hsb[j] = nullptr; }
    for (int j = 0; j < ns; ++j) L[i].hsw[j] = F(ip++);
    for (int j = 0; j < ns; ++j) L[i].hsb[j] = F(ip++);
    L[i].h_g = F(ip++); L[i].h_bb = F(ip++);
    L[i].v2h_w = F(ip++); L[i].v2h_b = F(ip++);
    for (int j = 0; j < 3; ++j) L[i].h2w[j] = F(ip++);
    for (int j = 0; j < 3; ++j) L[i].h2b[j] = F(ip++);
  }
  const float *o1w[3], *o1b[3], *o2w[3], *o2b[3];
  for (int j = 0; j < 3; ++j) o1w[j] = F(ip++);
  for (int j = 0; j < 3; ++j) o1b[j] = F(ip++);
  for (int j = 0; j < 3; ++j) o2w[j] = F(ip++);
  for (int j = 0; j < 3; ++j) o2b[j] = F(ip++);

  // ---- workspace layout --------------------------------------------------
  char* base = (char*)d_ws;
  size_t off = 0;
  auto alloc = [&](size_t bytes) -> void* {
    void* p = base + off;
    off += (bytes + 255) & ~(size_t)255;
    return p;
  };
  float* VXf   = (float*)alloc((size_t)S_TOT * CS * 4);
  float* HXf   = (float*)alloc((size_t)S_TOT * CS * 4);
  float* SKIPf = (float*)alloc((size_t)S_TOT * CS * 4);
  unsigned short* VXbf   = (unsigned short*)alloc((size_t)S_TOT * CS * 2);
  unsigned short* HXbf   = (unsigned short*)alloc((size_t)S_TOT * CS * 2);
  unsigned short* VCXbf  = (unsigned short*)alloc((size_t)S_TOT * CS * 2);
  unsigned short* Tbf    = (unsigned short*)alloc((size_t)S_TOT * CS * 2);
  unsigned short* HACTbf = (unsigned short*)alloc((size_t)S_TOT * CS * 2);
  unsigned short* SKIPbf = (unsigned short*)alloc((size_t)S_TOT * CS * 2);
  unsigned short* Pv0 = (unsigned short*)alloc((size_t)S_TOT * 32 * 2);
  unsigned short* Ph0 = (unsigned short*)alloc((size_t)S_TOT * 32 * 2);
  unsigned short* Y1  = (unsigned short*)alloc((size_t)S_TOT * 384 * 2);
  unsigned short *Wv[15], *Wh[15], *W2h[15];
  float *BH[15], *B2H[15];
  for (int i = 0; i < 15; ++i) {
    Wv[i]  = (unsigned short*)alloc((size_t)144 * (i == 0 ? 32 : 960) * 2);
    Wh[i]  = (unsigned short*)alloc((size_t)144 * (i == 0 ? 192 : 480) * 2);
    W2h[i] = (unsigned short*)alloc((size_t)144 * 160 * 2);
    BH[i]  = (float*)alloc(144 * 4);
    B2H[i] = (float*)alloc(144 * 4);
  }
  unsigned short* Wo1 = (unsigned short*)alloc((size_t)384 * 160 * 2);
  float* Bo1 = (float*)alloc(384 * 4);
  unsigned short* Wo2 = (unsigned short*)alloc((size_t)3 * 384 * 2);
  float* Bo2 = (float*)alloc(16);
  float* stats = (float*)alloc(4 * 160 * 4);   // sumV, sqV, sumH, sqH
  float* sumV = stats, *sqV = stats + 160, *sumH = stats + 320, *sqH = stats + 480;
  float* scV = (float*)alloc(160 * 4);
  float* biV = (float*)alloc(160 * 4);
  float* scH = (float*)alloc(160 * 4);
  float* biH = (float*)alloc(160 * 4);
  size_t used = off;

  // Zero entire used region: gives zero pads, zero-init HXf/SKIPf, and the
  // implicit zeros of the block-triangular packed weights.
  hipMemsetAsync(d_ws, 0, used, stream);

  auto cdiv = [](long long a, long long b) { return (int)((a + b - 1) / b); };

  // ---- pack all weights --------------------------------------------------
  for (int i = 0; i < 15; ++i) {
    if (i == 0) {
      k_pack_vconv<<<cdiv(144 * 6 * 3, 256), 256, 0, stream>>>(L[0].v_w, Wv[0], 3, 3, 32);
      k_pack_dense<<<cdiv(144 * 3, 256), 256, 0, stream>>>(L[0].h_w, Wh[0], 144, 3, 192, 0);
      k_pack_block<<<cdiv(96 * 2, 256), 256, 0, stream>>>(
          L[0].hsw[0], L[0].hsw[1], nullptr, Wh[0] + (size_t)48 * 192, 48, 1, 2, 192, 3);
      k_pack_dense<<<cdiv(144 * 144, 256), 256, 0, stream>>>(L[0].v2h_w, Wh[0], 144, 144, 192, 32);
      k_bias_h<<<1, 160, 0, stream>>>(L[0].h_b, L[0].v2h_b, nullptr, L[0].hsb[0], L[0].hsb[1], BH[0]);
    } else {
      k_pack_vconv<<<cdiv(144 * 6 * 144, 256), 256, 0, stream>>>(L[i].v_w, Wv[i], 144, 160, 960);
      k_pack_dense<<<cdiv(144 * 144, 256), 256, 0, stream>>>(L[i].h_w, Wh[i], 144, 144, 480, 0);
      k_pack_block<<<cdiv(144 * 144, 256), 256, 0, stream>>>(
          L[i].hsw[0], L[i].hsw[1], L[i].hsw[2], Wh[i], 48, 48, 3, 480, 160);
      k_pack_dense<<<cdiv(144 * 144, 256), 256, 0, stream>>>(L[i].v2h_w, Wh[i], 144, 144, 480, 320);
      k_bias_h<<<1, 160, 0, stream>>>(L[i].h_b, L[i].v2h_b, L[i].hsb[0], L[i].hsb[1], L[i].hsb[2], BH[i]);
    }
    k_pack_block<<<cdiv(144 * 144, 256), 256, 0, stream>>>(
        L[i].h2w[0], L[i].h2w[1], L[i].h2w[2], W2h[i], 48, 48, 3, 160, 0);
    k_bias_block<<<1, 256, 0, stream>>>(L[i].h2b[0], L[i].h2b[1], L[i].h2b[2], B2H[i], 48, 144);
  }
  k_pack_block<<<cdiv(384 * 144, 256), 256, 0, stream>>>(o1w[0], o1w[1], o1w[2], Wo1, 128, 48, 3, 160, 0);
  k_bias_block<<<cdiv(384, 256), 256, 0, stream>>>(o1b[0], o1b[1], o1b[2], Bo1, 128, 384);
  k_pack_block<<<cdiv(3 * 384, 256), 256, 0, stream>>>(o2w[0], o2w[1], o2w[2], Wo2, 1, 128, 3, 384, 0);
  k_bias_block<<<1, 256, 0, stream>>>(o2b[0], o2b[1], o2b[2], Bo2, 1, 3);

  // ---- layer-0 activation prep ------------------------------------------
  k_prep0<<<cdiv(S_TOT, 256), 256, 0, stream>>>(Xin, Pv0, Ph0);
  k_resvx<<<cdiv((long long)S_TOT * CS, 256), 256, 0, stream>>>(Xin, vx2vx_w, vx2vx_b, VXf);

  const float inv_n = 1.0f / (float)S_TOT;
  const dim3 gStats(S_TOT / 256, 72);

  // ---- layers ------------------------------------------------------------
  for (int i = 0; i < 15; ++i) {
    const int d = (i < 5) ? 1 : (i < 10) ? 2 : 4;
    hipMemsetAsync(stats, 0, 4 * 160 * 4, stream);

    // vertical stack GEMM -> VCXbf
    GemmCfg cv = {};
    cv.W = Wv[i]; cv.bias = L[i].v_b; cv.outbf = VCXbf; cv.ostride = CS;
    if (i == 0) {
      cv.Ktot = 32; cv.npanels = 1;
      cv.p[0] = {Pv0, 0, 0, 32, 32};
    } else {
      cv.Ktot = 960; cv.npanels = 6;
      for (int t = 0; t < 6; ++t)
        cv.p[t] = {VXbf, (t / 3 - 1) * d, (t % 3 - 1) * d, 160, CS};
    }
    k_gemm<<<dim3(S_TOT / 128, 3), 128, 0, stream>>>(cv);
    k_stats<<<gStats, 256, 0, stream>>>(VCXbf, CS, sumV, sqV);

    // horizontal stack GEMM (h_w on shifted hx + causal subpixel + v2h(vcx))
    GemmCfg chh = {};
    chh.W = Wh[i]; chh.bias = BH[i]; chh.outbf = Tbf; chh.ostride = CS;
    if (i == 0) {
      chh.Ktot = 192; chh.npanels = 2;
      chh.p[0] = {Ph0, 0, 0, 32, 32};
      chh.p[1] = {VCXbf, 0, 0, 160, CS};
    } else {
      chh.Ktot = 480; chh.npanels = 3;
      chh.p[0] = {HXbf, 0, -d, 160, CS};
      chh.p[1] = {HXbf, 0, 0, 160, CS};
      chh.p[2] = {VCXbf, 0, 0, 160, CS};
    }
    k_gemm<<<dim3(S_TOT / 128, 3), 128, 0, stream>>>(chh);
    k_stats<<<gStats, 256, 0, stream>>>(Tbf, CS, sumH, sqH);

    k_bnfin<<<1, 160, 0, stream>>>(sumV, sqV, L[i].v_g, L[i].v_bb, scV, biV, inv_n);
    k_bnfin<<<1, 160, 0, stream>>>(sumH, sqH, L[i].h_g, L[i].h_bb, scH, biH, inv_n);

    k_vupdate<<<cdiv((long long)S_TOT * 72, 256), 256, 0, stream>>>(VCXbf, scV, biV, VXf, VXbf);
    k_hupdate<<<cdiv((long long)S_TOT * 72, 256), 256, 0, stream>>>(Tbf, scH, biH, SKIPf, HACTbf);

    // hx = res_hx + h2h-subpixel(h_act): accumulate into HXf, mirror to HXbf
    GemmCfg c2 = {};
    c2.W = W2h[i]; c2.bias = B2H[i]; c2.accf = HXf; c2.accbf = HXbf;
    c2.ostride = CS; c2.Ktot = 160; c2.npanels = 1;
    c2.p[0] = {HACTbf, 0, 0, 160, CS};
    k_gemm<<<dim3(S_TOT / 128, 3), 128, 0, stream>>>(c2);
  }

  // ---- output head -------------------------------------------------------
  k_skipmm<<<cdiv((long long)S_TOT * 72, 256), 256, 0, stream>>>(SKIPf, SKIPbf);
  GemmCfg co = {};
  co.W = Wo1; co.bias = Bo1; co.outbf = Y1; co.ostride = 384;
  co.Ktot = 160; co.npanels = 1;
  co.p[0] = {SKIPbf, 0, 0, 160, CS};
  k_gemm<<<dim3(S_TOT / 128, 8), 128, 0, stream>>>(co);
  k_out2<<<cdiv(S_TOT, 256), 256, 0, stream>>>(Y1, Wo2, Bo2, (float*)d_out);
}
